// SpectralRecurrentCell_52974126629442
// MI455X (gfx1250) — matrix operand order repaired and checked by reading, not compiled, not run on hardware
//
#include <hip/hip_runtime.h>
#include <cstdint>
#include <cstddef>

// ---------------------------------------------------------------------------
// SpectralRecurrentCell for MI455X (gfx1250, wave32, WMMA + TDM)
//   B=64, T=1024, D_IN=1024, D_STATE=1024, N_PAIRS=512
//
//   (1) fp32 -> bf16 conversion of x + weights (workspace)
//   (2) x_proj = x @ W_in^T + b_in        : big WMMA GEMM, 2-D blocked
//   (3) scan: 4 WGs x 16 batch rows, independent; per step:
//         TDM async-loads x_{t+1} tile into LDS double buffer (tensorcnt)
//         GEMM1 [16,2048]x[2048,1024] (K split h|x) -> GELU -> hid
//         GEMM2 [16,1024]x[1024,512] -> tanh -> rotation (+x_proj) -> LN
//   (4) out = h_all @ W_out^T + b_out     : big WMMA GEMM, 2-D blocked
// ---------------------------------------------------------------------------

typedef __attribute__((ext_vector_type(16))) __bf16 v16bf;
typedef __attribute__((ext_vector_type(8)))  float  v8f;
typedef __attribute__((ext_vector_type(4)))  unsigned int u32x4;
typedef __attribute__((ext_vector_type(8)))  int    i32x8;
typedef __attribute__((ext_vector_type(4)))  int    i32x4;

#if defined(__has_builtin)
#  if __has_builtin(__builtin_amdgcn_tensor_load_to_lds)
#    define HAVE_TDM 1
#  else
#    define HAVE_TDM 0
#  endif
#else
#  define HAVE_TDM 0
#endif

union Frag16 { uint4 q[2]; v16bf v; };

// A fragment (16x32 bf16, row-major, stride in elems). Lane L (m=L&15, hi=L>>4)
// holds K = [hi*8, hi*8+8) and [16+hi*8, 16+hi*8+8).
__device__ __forceinline__ v16bf load_fragA(const __bf16* base, int rowBase,
                                            int stride, int k0, int lane) {
  const int m = lane & 15, hi = lane >> 4;
  const __bf16* p = base + (size_t)(rowBase + m) * stride + k0 + hi * 8;
  Frag16 f;
  f.q[0] = *(const uint4*)(p);
  f.q[1] = *(const uint4*)(p + 16);
  return f.v;
}

// B fragment (32x16 bf16): B[k][n] = W[n][k], W row-major [N][K].
// Lane L (n=L&15, hi=L>>4) holds K = [hi*8, hi*8+8) and [16+hi*8, 16+hi*8+8) from row n.
__device__ __forceinline__ v16bf load_fragB(const __bf16* W, int nBase,
                                            int stride, int k0, int lane) {
  const int n = lane & 15, hi = lane >> 4;
  const __bf16* p = W + (size_t)(nBase + n) * stride + k0 + hi * 8;
  Frag16 f;
  f.q[0] = *(const uint4*)(p);
  f.q[1] = *(const uint4*)(p + 16);
  return f.v;
}

__device__ __forceinline__ v8f wmma_bf16(v16bf a, v16bf b, v8f c) {
  return __builtin_amdgcn_wmma_f32_16x16x32_bf16(
      false, a, false, b, (short)0, c, false, false);
}

// TDM: async-load a 2-D tile (16 rows x 1024 bf16, row stride 1024*1024 elems)
// from global into LDS at byte offset ldsByteOff. ISA 08_async_tensor §8.3-8.4.
__device__ __forceinline__ void tdm_load_x_tile(const __bf16* gsrc,
                                                unsigned ldsByteOff) {
#if HAVE_TDM
  const unsigned long long ga = (unsigned long long)(uintptr_t)gsrc;
  u32x4 g0;
  g0[0] = 1u;                                        // count=1 (valid, user)
  g0[1] = ldsByteOff;                                // lds_addr (bytes)
  g0[2] = (unsigned)(ga & 0xffffffffu);              // global_addr lo
  g0[3] = (unsigned)((ga >> 32) & 0x01ffffffu)       // global_addr[56:32]
          | (2u << 30);                              // type=2 ("image")
  i32x8 g1;
  g1[0] = (int)(1u << 16);          // wg_mask=0 | data_size=1 (2 bytes)
  g1[1] = (int)(1024u << 16);       // atomic_bar=0 | tensor_dim0 lo = 1024
  g1[2] = (int)(16u << 16);         // tensor_dim0 hi=0 | tensor_dim1 lo = 16
  g1[3] = (int)(1024u << 16);       // tensor_dim1 hi=0 | tile_dim0 = 1024
  g1[4] = 16;                       // tile_dim1 = 16 | tile_dim2 = 0
  g1[5] = (int)(1024u * 1024u);     // tensor_dim0_stride lo (T*D elems)
  g1[6] = 0;                        // stride hi | tensor_dim1_stride lo
  g1[7] = 0;
  i32x4 gz = {0, 0, 0, 0};          // groups 2/3 zero (2-D tensor)
#  if __clang_major__ >= 23
  i32x8 gz8 = {0, 0, 0, 0, 0, 0, 0, 0};
  __builtin_amdgcn_tensor_load_to_lds(g0, g1, gz, gz, gz8, 0);
#  else
  __builtin_amdgcn_tensor_load_to_lds(g0, g1, gz, gz, 0);
#  endif
#else
  (void)gsrc; (void)ldsByteOff;
#endif
}

// ---------------------------------------------------------------------------
__global__ void __launch_bounds__(256) cvt_f32_bf16(const float* __restrict__ in,
                                                    __bf16* __restrict__ out,
                                                    size_t n) {
  size_t i = (size_t)blockIdx.x * 256 + threadIdx.x;
  const size_t stride = (size_t)gridDim.x * 256;
  for (; i < n; i += stride) out[i] = (__bf16)in[i];
}

// ---------------------------------------------------------------------------
// GEMM: out[M,N] = A[M,K](bf16) @ W[N,K]^T(bf16) + bias, fp32 out. N == 1024.
// 2-D grid: block = 64 rows x 256 cols; wave = 1 M-tile x 8 N-tiles (4x2).
// Cuts W re-reads ~1000x vs 1-D blocking (weights stay L2-resident).
// ---------------------------------------------------------------------------
__global__ void __launch_bounds__(256) gemm_bf16_f32(
    const __bf16* __restrict__ A, const __bf16* __restrict__ W,
    const float* __restrict__ bias, float* __restrict__ out, int N, int K) {
  const int lane = threadIdx.x & 31;
  const int w = threadIdx.x >> 5;
  const int mRow = (blockIdx.x * 4 + (w >> 1)) << 4;
  const int nt0 = blockIdx.y * 16 + (w & 1) * 8;
  v8f acc[8];
#pragma unroll
  for (int i = 0; i < 8; ++i) { v8f z = {}; acc[i] = z; }
  for (int s = 0; s < K; s += 32) {
    v16bf a = load_fragA(A, mRow, K, s, lane);
#pragma unroll
    for (int i = 0; i < 8; ++i) {
      v16bf b = load_fragB(W, (nt0 + i) << 4, K, s, lane);
      acc[i] = wmma_bf16(a, b, acc[i]);
    }
  }
  const int hi = lane >> 4, nl = lane & 15;
#pragma unroll
  for (int i = 0; i < 8; ++i) {
    const int n = ((nt0 + i) << 4) + nl;
    const float bv = bias[n];
#pragma unroll
    for (int r = 0; r < 8; ++r) {
      out[(size_t)(mRow + hi * 8 + r) * N + n] = acc[i][r] + bv;
    }
  }
}

// ---------------------------------------------------------------------------
// Recurrent scan. 4 blocks x 256 threads; block = batch rows [16*bid, +16).
// Dynamic LDS (192KB): hS f32 [16][1024] | hbS bf16 [16][1024] |
//                      hidS bf16 [16][1024] | xS bf16 [2][16][1024]
// ---------------------------------------------------------------------------
#define LDS_HS    0
#define LDS_HBS   (16 * 1024 * 4)
#define LDS_HIDS  (LDS_HBS + 16 * 1024 * 2)
#define LDS_XS    (LDS_HIDS + 16 * 1024 * 2)
#define LDS_TOTAL (LDS_XS + 2 * 16 * 1024 * 2)

__global__ void __launch_bounds__(256) scan_kernel(
    const float* __restrict__ h0, const __bf16* __restrict__ xb,
    const float* __restrict__ xproj,
    const __bf16* __restrict__ Wr1, const float* __restrict__ br1,
    const __bf16* __restrict__ Wr2, const float* __restrict__ br2,
    const float* __restrict__ lng, const float* __restrict__ lnb,
    __bf16* __restrict__ hall, float* __restrict__ hfinal) {
  extern __shared__ __align__(16) char smem[];
  float*  hS   = (float*)(smem + LDS_HS);     // [16][1024]
  __bf16* hbS  = (__bf16*)(smem + LDS_HBS);   // [16][1024]
  __bf16* hidS = (__bf16*)(smem + LDS_HIDS);  // [16][1024]
  __bf16* xS   = (__bf16*)(smem + LDS_XS);    // [2][16][1024]

  const int tid = threadIdx.x;
  const int lane = tid & 31, w = tid >> 5;
  const int b0 = blockIdx.x << 4;
  const float EIG_STEP = 0.099f / 1023.0f;
  const unsigned ldsXBase = __builtin_amdgcn_groupstaticsize() + (unsigned)LDS_XS;

  for (int i = tid; i < 16 * 1024; i += 256)
    hS[i] = h0[(size_t)(b0 + (i >> 10)) * 1024 + (i & 1023)];

#if HAVE_TDM
  if (w == 0)  // prologue: async-fetch x_0 into buffer 0
    tdm_load_x_tile(xb + (size_t)b0 * 1024 * 1024, ldsXBase);
#endif
  __syncthreads();

  for (int t = 0; t < 1024; ++t) {
#if HAVE_TDM
    const __bf16* xcur = xS + (t & 1) * 16 * 1024;
    if (w == 0 && t + 1 < 1024)  // async-fetch x_{t+1} into other buffer
      tdm_load_x_tile(xb + ((size_t)b0 * 1024 + t + 1) * 1024,
                      ldsXBase + (unsigned)(((t + 1) & 1) * 32768));
#else
    const __bf16* xcur = xS;
    for (int i = tid; i < 2048; i += 256) {  // 16 rows x 128 chunks of 8 bf16
      const int r = i >> 7, c8 = (i & 127) << 3;
      *(uint4*)(xS + r * 1024 + c8) =
          *(const uint4*)(xb + ((size_t)(b0 + r) * 1024 + t) * 1024 + c8);
    }
#endif
    // h -> bf16 copy (overlaps with TDM in flight)
    for (int i = tid; i < 16 * 1024; i += 256) hbS[i] = (__bf16)hS[i];
    // prefetch next step's x_proj rows (global_prefetch_b8)
    if (t + 1 < 1024) {
      for (int i = tid; i < 1024; i += 256)
        __builtin_prefetch(
            xproj + ((size_t)(b0 + (i >> 6)) * 1024 + t + 1) * 1024 +
                (size_t)(i & 63) * 16, 0, 1);
    }
#if HAVE_TDM
    if (w == 0) {  // retire x_t's DMA (in-order per wave); x_{t+1} stays in flight
      if (t + 1 < 1024) __builtin_amdgcn_s_wait_tensorcnt(1);
      else              __builtin_amdgcn_s_wait_tensorcnt(0);
    }
#endif
    __syncthreads();

    // GEMM1: hid = GELU([h|x] @ Wr1^T + br1), K=2048 split h-half / x-half
    {
      v8f acc[8];
#pragma unroll
      for (int i = 0; i < 8; ++i) { v8f z = {}; acc[i] = z; }
      for (int s = 0; s < 1024; s += 32) {  // K-half 1: h
        v16bf a = load_fragA(hbS, 0, 1024, s, lane);
#pragma unroll
        for (int i = 0; i < 8; ++i) {
          v16bf b = load_fragB(Wr1, (w * 8 + i) << 4, 2048, s, lane);
          acc[i] = wmma_bf16(a, b, acc[i]);
        }
      }
      for (int s = 0; s < 1024; s += 32) {  // K-half 2: x_t
        v16bf a = load_fragA(xcur, 0, 1024, s, lane);
#pragma unroll
        for (int i = 0; i < 8; ++i) {
          v16bf b = load_fragB(Wr1, (w * 8 + i) << 4, 2048, 1024 + s, lane);
          acc[i] = wmma_bf16(a, b, acc[i]);
        }
      }
      const int hi = lane >> 4, nl = lane & 15;
#pragma unroll
      for (int i = 0; i < 8; ++i) {
        const int n = ((w * 8 + i) << 4) + nl;
        const float bv = br1[n];
#pragma unroll
        for (int r = 0; r < 8; ++r) {
          const float xv = acc[i][r] + bv;
          const float g = 0.5f * xv * (1.0f + erff(xv * 0.70710678f));
          hidS[(hi * 8 + r) * 1024 + n] = (__bf16)g;
        }
      }
    }
    __syncthreads();

    // GEMM2: angles = tanh(hid @ Wr2^T + br2)*0.1pi; rotate + add x_proj
    {
      v8f acc[4];
#pragma unroll
      for (int i = 0; i < 4; ++i) { v8f z = {}; acc[i] = z; }
      for (int s = 0; s < 1024; s += 32) {
        v16bf a = load_fragA(hidS, 0, 1024, s, lane);
#pragma unroll
        for (int i = 0; i < 4; ++i) {
          v16bf b = load_fragB(Wr2, (w * 4 + i) << 4, 1024, s, lane);
          acc[i] = wmma_bf16(a, b, acc[i]);
        }
      }
      const int hi = lane >> 4, nl = lane & 15;
#pragma unroll
      for (int i = 0; i < 4; ++i) {
        const int p = ((w * 4 + i) << 4) + nl;
        const float bv = br2[p];
#pragma unroll
        for (int r = 0; r < 8; ++r) {
          const int m = hi * 8 + r;
          const float ang = tanhf(acc[i][r] + bv) * 0.31415926535f;
          float sn, c;
          sincosf(ang, &sn, &c);
          const int i0 = p << 1;
          const float e0 = hS[m * 1024 + i0]     * (0.9f + EIG_STEP * i0);
          const float e1 = hS[m * 1024 + i0 + 1] * (0.9f + EIG_STEP * (i0 + 1));
          const float* xp = xproj + ((size_t)(b0 + m) * 1024 + t) * 1024 + i0;
          hS[m * 1024 + i0]     = c * e0 - sn * e1 + xp[0];
          hS[m * 1024 + i0 + 1] = sn * e0 + c * e1 + xp[1];
        }
      }
    }
    __syncthreads();

    // LayerNorm: wave w handles rows 2w, 2w+1; store h_new bf16 for GEMM (4)
#pragma unroll
    for (int rr = 0; rr < 2; ++rr) {
      const int r = w * 2 + rr;
      float sum = 0.f, sq = 0.f;
#pragma unroll
      for (int j = 0; j < 32; ++j) {
        const float v = hS[r * 1024 + lane + (j << 5)];
        sum += v; sq += v * v;
      }
#pragma unroll
      for (int off = 16; off > 0; off >>= 1) {
        sum += __shfl_xor(sum, off, 32);
        sq  += __shfl_xor(sq, off, 32);
      }
      const float mu = sum * (1.0f / 1024.0f);
      const float var = sq * (1.0f / 1024.0f) - mu * mu;
      const float rs = rsqrtf(var + 1e-5f);
#pragma unroll
      for (int j = 0; j < 32; ++j) {
        const int i = lane + (j << 5);
        const float y = (hS[r * 1024 + i] - mu) * rs * lng[i] + lnb[i];
        hS[r * 1024 + i] = y;
        hall[((size_t)(b0 + r) * 1024 + t) * 1024 + i] = (__bf16)y;
      }
    }
    __syncthreads();
  }

  for (int i = tid; i < 16 * 1024; i += 256)
    hfinal[(size_t)(b0 + (i >> 10)) * 1024 + (i & 1023)] = hS[i];
}

// ---------------------------------------------------------------------------
extern "C" void kernel_launch(void* const* d_in, const int* in_sizes, int n_in,
                              void* d_out, int out_size, void* d_ws,
                              size_t ws_size, hipStream_t stream) {
  (void)in_sizes; (void)n_in; (void)out_size; (void)ws_size;
  const float* x     = (const float*)d_in[0];
  const float* h0    = (const float*)d_in[1];
  const float* W_in  = (const float*)d_in[2];
  const float* b_in  = (const float*)d_in[3];
  const float* W_r1  = (const float*)d_in[4];
  const float* b_r1  = (const float*)d_in[5];
  const float* W_r2  = (const float*)d_in[6];
  const float* b_r2  = (const float*)d_in[7];
  const float* W_out = (const float*)d_in[8];
  const float* b_out = (const float*)d_in[9];
  const float* ln_g  = (const float*)d_in[10];
  const float* ln_b  = (const float*)d_in[11];

  float* out = (float*)d_out;
  float* hfinal = out + (size_t)64 * 1024 * 1024;

  char* ws = (char*)d_ws;
  size_t off = 0;
  __bf16* xb    = (__bf16*)(ws + off); off += (size_t)64 * 1024 * 1024 * 2;
  __bf16* Winb  = (__bf16*)(ws + off); off += (size_t)1024 * 1024 * 2;
  __bf16* Wr1b  = (__bf16*)(ws + off); off += (size_t)1024 * 2048 * 2;
  __bf16* Wr2b  = (__bf16*)(ws + off); off += (size_t)512 * 1024 * 2;
  __bf16* Woutb = (__bf16*)(ws + off); off += (size_t)1024 * 1024 * 2;
  float*  xproj = (float*)(ws + off);  off += (size_t)64 * 1024 * 1024 * 4;
  __bf16* hall  = (__bf16*)(ws + off); off += (size_t)64 * 1024 * 1024 * 2;

  cvt_f32_bf16<<<4096, 256, 0, stream>>>(x, xb, (size_t)64 * 1024 * 1024);
  cvt_f32_bf16<<<512, 256, 0, stream>>>(W_in, Winb, (size_t)1024 * 1024);
  cvt_f32_bf16<<<512, 256, 0, stream>>>(W_r1, Wr1b, (size_t)1024 * 2048);
  cvt_f32_bf16<<<512, 256, 0, stream>>>(W_r2, Wr2b, (size_t)512 * 1024);
  cvt_f32_bf16<<<512, 256, 0, stream>>>(W_out, Woutb, (size_t)1024 * 1024);

  dim3 gemmGrid(1024, 4);
  gemm_bf16_f32<<<gemmGrid, 256, 0, stream>>>(xb, Winb, b_in, xproj, 1024, 1024);

  scan_kernel<<<4, 256, LDS_TOTAL, stream>>>(h0, xb, xproj, Wr1b, b_r1,
                                             Wr2b, b_r2, ln_g, ln_b, hall,
                                             hfinal);

  gemm_bf16_f32<<<gemmGrid, 256, 0, stream>>>(hall, Woutb, b_out, out, 1024, 1024);
}
